// IPEXTransformerQLinear_33122787786863
// MI455X (gfx1250) — matrix operand order repaired and checked by reading, not compiled, not run on hardware
//
#include <hip/hip_runtime.h>
#include <cstdint>
#include <cstddef>

#define MDIM 8192
#define KDIM 4096
#define NDIM 11008
#define QGS  128

// Tiling: 256(M) x 256(N) x 128(K) per workgroup, 512 threads = 16 wave32s (4x4),
// each wave owns a 64x64 sub-tile = 4x4 WMMA tiles of 16x16, f32 accum.

typedef __attribute__((ext_vector_type(16))) _Float16 v16h;
typedef __attribute__((ext_vector_type(2)))  _Float16 h2;
typedef __attribute__((ext_vector_type(8)))  float    v8f;

union Frag { v16h v; uint4 q[2]; };

__global__ __launch_bounds__(512)
void w4a16_wmma_gemm(const float* __restrict__ A,
                     const int*   __restrict__ Wq,
                     const float* __restrict__ scale,
                     const int*   __restrict__ zp,
                     float*       __restrict__ out)
{
  // fp16 pairs packed in uint: lA[m][k/2] (row-major K), lB[n][k/2] (B transposed)
  __shared__ unsigned lA[256 * 64];   // 64 KB
  __shared__ unsigned lB[256 * 64];   // 64 KB

  const int tid  = threadIdx.x;
  const int n0   = blockIdx.x * 256;
  const int m0   = blockIdx.y * 256;

  const int lane = tid & 31;
  const int wave = tid >> 5;        // 0..15
  const int wm   = wave >> 2;       // 0..3 : 64-row band
  const int wn   = wave & 3;        // 0..3 : 64-col band
  const int half = lane >> 4;       // WMMA lane-half
  const int lrow = lane & 15;

  // loader coordinates (fixed column group per thread -> coalesced)
  const int acol4 = tid & 31;       // float4 index within a 128-float A row
  const int arow0 = tid >> 5;       // A row base, step 16
  const int bcol4 = tid & 63;       // int4 index within a 256-int W row
  const int brow0 = tid >> 6;       // packed-K row base, step 8
  const int nloc  = bcol4 * 4;      // 4 consecutive n per thread

  v8f acc[4][4] = {};

  for (int kt = 0; kt < KDIM / 128; ++kt) {
    const int k0 = kt * 128;

    // group == k-tile (QGS == BLOCK_K): per-thread dequant constants, 4 columns
    const float4 sc = *(const float4*)(scale + (size_t)kt * NDIM + n0 + nloc);
    const int4   zq = *(const int4*)  (zp    + (size_t)kt * NDIM + n0 + nloc);
    const float sv[4] = { sc.x, sc.y, sc.z, sc.w };
    const int   zv[4] = { zq.x, zq.y, zq.z, zq.w };
    h2 s_pk[4], d_pk[4];
    #pragma unroll
    for (int j = 0; j < 4; ++j) {
      const _Float16 hs = (_Float16)sv[j];
      const _Float16 hd = (_Float16)(-(1024.0f + (float)zv[j]));
      s_pk[j] = (h2){hs, hs};
      d_pk[j] = (h2){hd, hd};
    }

    // ---- stage A tile: fp32 -> fp16 pairs (v_cvt_pkrtz_f16_f32) ----
    #pragma unroll
    for (int i = 0; i < 16; ++i) {
      const int row = arow0 + i * 16;
      const float4 v = *(const float4*)(A + (size_t)(m0 + row) * KDIM + k0 + acol4 * 4);
      const h2 p0 = __builtin_bit_cast(h2, __builtin_amdgcn_cvt_pkrtz(v.x, v.y));
      const h2 p1 = __builtin_bit_cast(h2, __builtin_amdgcn_cvt_pkrtz(v.z, v.w));
      uint2 p;
      p.x = __builtin_bit_cast(unsigned, p0);
      p.y = __builtin_bit_cast(unsigned, p1);
      *(uint2*)&lA[row * 64 + acol4 * 2] = p;
    }

    // ---- stage B tile: nibble->fp16 via exponent-bias magic, transposed [n][k/2] ----
    // byte b = hi<<4 | lo ; 0x6400|nib == 1024+nib (exact in fp16)
    // (1024+nib) + (-(1024+zp)) is exact; one rounding in the scale multiply.
    #pragma unroll
    for (int i = 0; i < 8; ++i) {
      const int prow = brow0 + i * 8;  // packed-K row 0..63 (k = 2*prow, 2*prow+1)
      const int4 w = *(const int4*)(Wq + (size_t)(kt * 64 + prow) * NDIM + n0 + nloc);
      const unsigned wv[4] = { (unsigned)w.x, (unsigned)w.y, (unsigned)w.z, (unsigned)w.w };
      #pragma unroll
      for (int j = 0; j < 4; ++j) {
        unsigned t = ((wv[j] << 12) | wv[j]) & 0x000F000Fu;
        t |= 0x64006400u;
        const h2 hp = __builtin_bit_cast(h2, t);
        const h2 r  = (hp + d_pk[j]) * s_pk[j];
        lB[(nloc + j) * 64 + prow] = __builtin_bit_cast(unsigned, r);
      }
    }

    // prefetch next tile while we compute this one
    if (kt + 1 < KDIM / 128) {
      __builtin_prefetch(A + (size_t)(m0 + arow0) * KDIM + k0 + 128, 0, 0);
      __builtin_prefetch(Wq + (size_t)((kt + 1) * 64 + brow0) * NDIM + n0 + nloc, 0, 0);
    }

    __syncthreads();

    // ---- 4 WMMA k-steps of K=32 : 16 v_wmma per step, 64 per k-tile ----
    #pragma unroll
    for (int ks = 0; ks < 4; ++ks) {
      const int ku = ks * 16;  // uint (fp16-pair) offset = k/2
      Frag a[4], b[4];
      #pragma unroll
      for (int tm = 0; tm < 4; ++tm) {
        // A 16x32 f16 layout: lanes0-15 K {0..7,16..23}, lanes16-31 +8
        const unsigned* base = &lA[(wm * 64 + tm * 16 + lrow) * 64 + ku + half * 4];
        a[tm].q[0] = *(const uint4*)(base);
        a[tm].q[1] = *(const uint4*)(base + 8);
      }
      #pragma unroll
      for (int tn = 0; tn < 4; ++tn) {
        // B 32x16 f16 layout: lane = N, lanes0-15 K {0..7,16..23}, lanes16-31 +8
        const unsigned* base = &lB[(wn * 64 + tn * 16 + lrow) * 64 + ku + half * 4];
        b[tn].q[0] = *(const uint4*)(base);
        b[tn].q[1] = *(const uint4*)(base + 8);
      }
      #pragma unroll
      for (int tm = 0; tm < 4; ++tm)
        #pragma unroll
        for (int tn = 0; tn < 4; ++tn)
          acc[tm][tn] = __builtin_amdgcn_wmma_f32_16x16x32_f16(
              false, a[tm].v, false, b[tn].v, (short)0, acc[tm][tn], false, false);
    }

    __syncthreads();
  }

  // ---- epilogue: C/D 16x16 f32 layout (VGPR r -> M=r / M=8+r per lane half) ----
  #pragma unroll
  for (int tm = 0; tm < 4; ++tm) {
    const int mbase = m0 + wm * 64 + tm * 16 + half * 8;
    #pragma unroll
    for (int tn = 0; tn < 4; ++tn) {
      const int n = n0 + wn * 64 + tn * 16 + lrow;
      #pragma unroll
      for (int r = 0; r < 8; ++r) {
        out[(size_t)(mbase + r) * NDIM + n] = acc[tm][tn][r];
      }
    }
  }
}

extern "C" void kernel_launch(void* const* d_in, const int* in_sizes, int n_in,
                              void* d_out, int out_size, void* d_ws, size_t ws_size,
                              hipStream_t stream) {
  (void)in_sizes; (void)n_in; (void)out_size; (void)d_ws; (void)ws_size;
  const float* A     = (const float*)d_in[0];
  const int*   Wq    = (const int*)d_in[1];
  const float* scale = (const float*)d_in[2];
  const int*   zp    = (const int*)d_in[3];
  float*       out   = (float*)d_out;

  dim3 grid(NDIM / 256, MDIM / 256);  // 43 x 32 workgroups
  w4a16_wmma_gemm<<<grid, 512, 0, stream>>>(A, Wq, scale, zp, out);
}
